// Model_36928128811716
// MI455X (gfx1250) — matrix operand order repaired and checked by reading, not compiled, not run on hardware
//
#include <hip/hip_runtime.h>

#define H_DIM 256
#define R_REL 3
#define NCL 20000
#define NDR 4000
#define FCL 3831
#define FCLP 3840   // FCL padded to multiple of 32
#define FDR 1024
#define NE  500000
#define NEL 200000

typedef __attribute__((ext_vector_type(16))) __bf16 v16bf;
typedef __attribute__((ext_vector_type(8)))  float  v8f;

static __device__ __forceinline__ __bf16 f2bf(float x) { return (__bf16)x; }

union Frag { v16bf v; uint4 q[2]; };

// ---------------------------------------------------------------------------
// Pack fp32 [M,K] -> bf16 [M,Kp] row-major, zero-padded K tail.
// ---------------------------------------------------------------------------
__global__ void pack_a(const float* __restrict__ A, __bf16* __restrict__ out,
                       int M, int K, int Kp)
{
    long long i = (long long)blockIdx.x * blockDim.x + threadIdx.x;
    if (i >= (long long)M * Kp) return;
    int m = (int)(i / Kp), k = (int)(i % Kp);
    out[i] = f2bf(k < K ? A[(long long)m * K + k] : 0.0f);
}

// Pack fp32 B [b][K,N] -> bf16 BT [b][N,Kp] (transposed, zero-padded).
__global__ void pack_bt(const float* __restrict__ B, __bf16* __restrict__ out,
                        int K, int N, int Kp)
{
    long long i = (long long)blockIdx.x * blockDim.x + threadIdx.x;
    long long tot = (long long)N * Kp;
    if (i >= tot) return;
    int b = blockIdx.y;
    int n = (int)(i / Kp), k = (int)(i % Kp);
    float v = (k < K) ? B[(long long)b * K * N + (long long)k * N + n] : 0.0f;
    out[(long long)b * tot + i] = f2bf(v);
}

// ---------------------------------------------------------------------------
// Fused 4-output GEMM: C[g] = A @ BT[g]^T for g=0..3 (3 relation weights +
// root weight sharing the same A). A [M,Kp] bf16, BT [4,256,Kp] bf16,
// C [4,M,256] fp32. Wave computes 16 rows x 64 cols for all 4 outputs:
// one A fragment feeds 16 WMMAs per K-step. All loads are aligned b128.
// ---------------------------------------------------------------------------
__global__ __launch_bounds__(128)
void wmma_gemm_r4(const __bf16* __restrict__ A, const __bf16* __restrict__ BT,
                  float* __restrict__ C, int M, int Kp)
{
    const int N = H_DIM;
    int wave = threadIdx.x >> 5;
    int lane = threadIdx.x & 31;
    int rsel = lane & 15;          // A: row in tile. B/C: column within 16-frag
    int hi   = lane >> 4;          // lane half (ISA 16-bit operand layout)

    int row0 = blockIdx.y * 64 + wave * 16;
    if (row0 >= M) return;         // M multiple of 16 for all our shapes
    int col0 = blockIdx.x * 64;

    const __bf16* arow = A + (size_t)(row0 + rsel) * Kp + hi * 8;
    const __bf16* brow = BT + (size_t)(col0 + rsel) * Kp + hi * 8;
    const size_t relS = (size_t)N * Kp;   // stride between the 4 B matrices
    const size_t colS = (size_t)16 * Kp;  // stride between column fragments

    v8f acc[4][4] = {};

    for (int k0 = 0; k0 < Kp; k0 += 32) {
        Frag af;
        const uint4* ap = (const uint4*)(arow + k0);
        af.q[0] = ap[0];           // K = k0 + hi*8 + 0..7
        af.q[1] = ap[2];           // K = k0 + 16 + hi*8 + 0..7
#pragma unroll
        for (int r = 0; r < 4; ++r) {
#pragma unroll
            for (int j = 0; j < 4; ++j) {
                Frag bf;
                const uint4* bp = (const uint4*)(brow + r * relS + j * colS + k0);
                bf.q[0] = bp[0];   // K = k0 + hi*8 + 0..7
                bf.q[1] = bp[2];   // K = k0 + 16 + hi*8 + 0..7
                acc[r][j] = __builtin_amdgcn_wmma_f32_16x16x32_bf16(
                    false, af.v, false, bf.v, (short)0, acc[r][j], false, false);
            }
        }
    }
    // C layout: VGPR v holds (M = v + 8*hi, N = rsel)
#pragma unroll
    for (int r = 0; r < 4; ++r)
#pragma unroll
        for (int j = 0; j < 4; ++j)
#pragma unroll
            for (int v = 0; v < 8; ++v)
                C[((size_t)r * M + row0 + hi * 8 + v) * N + col0 + 16 * j + rsel]
                    = acc[r][j][v];
}

// ---------------------------------------------------------------------------
// Edge aggregation
// ---------------------------------------------------------------------------
__global__ void edge_count(const int* __restrict__ dst, const int* __restrict__ et,
                           float* __restrict__ cnt, int E)
{
    int e = blockIdx.x * blockDim.x + threadIdx.x;
    if (e < E) atomicAdd(&cnt[dst[e] * R_REL + et[e]], 1.0f);
}

// 64 threads per edge, 4 channels each (H=256). h is [R(,+root), n_src, H].
__global__ void edge_scatter(const float* __restrict__ h, const int* __restrict__ src,
                             const int* __restrict__ dst, const int* __restrict__ et,
                             float* __restrict__ sums, int E, int n_src)
{
    long long t = (long long)blockIdx.x * blockDim.x + threadIdx.x;
    long long e = t >> 6;
    if (e >= E) return;
    int c = ((int)t & 63) * 4;
    int r = et[e], s = src[e], d = dst[e];
    const float4 m = *(const float4*)(h + ((long long)r * n_src + s) * H_DIM + c);
    float* o = sums + ((long long)d * R_REL + r) * H_DIM + c;
    atomicAdd(o + 0, m.x); atomicAdd(o + 1, m.y);
    atomicAdd(o + 2, m.z); atomicAdd(o + 3, m.w);
}

// z = act( sum_r sums[n,r,:]/max(cnt,1) + root[n,:] + bias ) -> bf16 (WMMA A)
__global__ void combine(const float* __restrict__ sums, const float* __restrict__ cnt,
                        const float* __restrict__ root, const float* __restrict__ bias,
                        __bf16* __restrict__ z, int n_dst, int relu)
{
    long long i = (long long)blockIdx.x * blockDim.x + threadIdx.x;
    if (i >= (long long)n_dst * H_DIM) return;
    int n = (int)(i >> 8);
    int c = (int)(i & 255);
    float a = 0.f;
#pragma unroll
    for (int r = 0; r < R_REL; ++r) {
        float cc = cnt[n * R_REL + r];
        a += sums[((long long)n * R_REL + r) * H_DIM + c] / fmaxf(cc, 1.0f);
    }
    a += root[i] + bias[c];
    if (relu) a = fmaxf(a, 0.f);
    z[i] = f2bf(a);
}

__global__ void init_out(float* __restrict__ out, const float* __restrict__ b, int n)
{
    int i = blockIdx.x * blockDim.x + threadIdx.x;
    if (i < n) out[i] = b[0];
}

// ---------------------------------------------------------------------------
// Fused decoder: A = gather-concat(z_cl2[row], z_dr2[col]) bf16 (16x512/tile),
// WMMA vs lin1_w^T (bf16 [256,512]), bias+ReLU in-register, dot with lin2_w,
// shfl_xor reduce, atomic into out (pre-initialized to lin2_b).
// ---------------------------------------------------------------------------
__global__ __launch_bounds__(128)
void decoder_wmma(const __bf16* __restrict__ zc, const __bf16* __restrict__ zd,
                  const int* __restrict__ row, const int* __restrict__ col,
                  const __bf16* __restrict__ w1t, const float* __restrict__ b1,
                  const float* __restrict__ w2, float* __restrict__ out)
{
    int wave = threadIdx.x >> 5;
    int lane = threadIdx.x & 31;
    int rsel = lane & 15;
    int hi   = lane >> 4;
    int e0 = blockIdx.y * 16;
    int n0 = (blockIdx.x * 4 + wave) * 16;

    int e = e0 + rsel;
    const __bf16* za = zc + (size_t)row[e] * H_DIM;
    const __bf16* zb = zd + (size_t)col[e] * H_DIM;
    const __bf16* brow = w1t + (size_t)(n0 + rsel) * (2 * H_DIM) + hi * 8;

    v8f acc = {};
    for (int k0 = 0; k0 < 2 * H_DIM; k0 += 32) {
        // fragments never straddle the 256 boundary (k0 multiple of 32)
        const __bf16* src = (k0 < H_DIM) ? (za + k0) : (zb + (k0 - H_DIM));
        Frag af, bf;
        af.q[0] = *(const uint4*)(src + hi * 8);
        af.q[1] = *(const uint4*)(src + hi * 8 + 16);
        const uint4* bp = (const uint4*)(brow + k0);
        bf.q[0] = bp[0];
        bf.q[1] = bp[2];
        acc = __builtin_amdgcn_wmma_f32_16x16x32_bf16(
            false, af.v, false, bf.v, (short)0, acc, false, false);
    }

    int n = n0 + rsel;
    float bias = b1[n], wv = w2[n];
    float p[8];
#pragma unroll
    for (int v = 0; v < 8; ++v) {
        float h = acc[v] + bias;
        h = h > 0.f ? h : 0.f;
        p[v] = h * wv;           // partial of out[e0 + v + 8*hi] over 16 cols
    }
#pragma unroll
    for (int m = 1; m < 16; m <<= 1) {
#pragma unroll
        for (int v = 0; v < 8; ++v) p[v] += __shfl_xor(p[v], m, 32);
    }
    if (rsel == 0) {
#pragma unroll
        for (int v = 0; v < 8; ++v) atomicAdd(&out[e0 + hi * 8 + v], p[v]);
    }
}

// ---------------------------------------------------------------------------
extern "C" void kernel_launch(void* const* d_in, const int* in_sizes, int n_in,
                              void* d_out, int out_size, void* d_ws, size_t ws_size,
                              hipStream_t stream)
{
    const float* x_cl  = (const float*)d_in[0];
    const float* x_dr  = (const float*)d_in[1];
    const int* src_sen = (const int*)d_in[2];
    const int* dst_sen = (const int*)d_in[3];
    const int* et_sen  = (const int*)d_in[4];
    const int* src_rev = (const int*)d_in[5];
    const int* dst_rev = (const int*)d_in[6];
    const int* et_rev  = (const int*)d_in[7];
    const int* row     = (const int*)d_in[8];
    const int* col     = (const int*)d_in[9];
    const float* W1s = (const float*)d_in[10];
    const float* r1s = (const float*)d_in[11];
    const float* b1s = (const float*)d_in[12];
    const float* W1r = (const float*)d_in[13];
    const float* r1r = (const float*)d_in[14];
    const float* b1r = (const float*)d_in[15];
    const float* W2s = (const float*)d_in[16];
    const float* r2s = (const float*)d_in[17];
    const float* b2s = (const float*)d_in[18];
    const float* W2r = (const float*)d_in[19];
    const float* r2r = (const float*)d_in[20];
    const float* b2r = (const float*)d_in[21];
    const float* l1w = (const float*)d_in[22];
    const float* l1b = (const float*)d_in[23];
    const float* l2w = (const float*)d_in[24];
    const float* l2b = (const float*)d_in[25];
    float* out = (float*)d_out;

    // ---- workspace carve (256B-aligned buffers) ----
    char* wp = (char*)d_ws;
    auto carve = [&](size_t bytes) -> char* {
        char* r = wp; wp += (bytes + 255) & ~(size_t)255; return r;
    };
    float* c_cl  = (float*)carve(sizeof(float) * 4ull * NCL * H_DIM); // slices 0-2: h, 3: root
    float* c_dr  = (float*)carve(sizeof(float) * 4ull * NDR * H_DIM);
    float* sm_cl = (float*)carve(sizeof(float) * (size_t)NCL * R_REL * H_DIM);
    float* sm_dr = (float*)carve(sizeof(float) * (size_t)NDR * R_REL * H_DIM);
    float* ct_cl = (float*)carve(sizeof(float) * (size_t)NCL * R_REL);
    float* ct_dr = (float*)carve(sizeof(float) * (size_t)NDR * R_REL);
    __bf16* xcl_bf = (__bf16*)carve(2ull * NCL * FCLP);
    __bf16* xdr_bf = (__bf16*)carve(2ull * NDR * FDR);
    __bf16* bt1_cl = (__bf16*)carve(2ull * 4 * H_DIM * FCLP);  // W1s(3) + root1_rev
    __bf16* bt1_dr = (__bf16*)carve(2ull * 4 * H_DIM * FDR);   // W1r(3) + root1_sen
    __bf16* bt2_cl = (__bf16*)carve(2ull * 4 * H_DIM * H_DIM); // W2s(3) + root2_rev
    __bf16* bt2_dr = (__bf16*)carve(2ull * 4 * H_DIM * H_DIM); // W2r(3) + root2_sen
    __bf16* l1t    = (__bf16*)carve(2ull * H_DIM * 2 * H_DIM); // lin1_w^T
    __bf16* zcl_bf  = (__bf16*)carve(2ull * NCL * H_DIM);
    __bf16* zdr_bf  = (__bf16*)carve(2ull * NDR * H_DIM);
    __bf16* zcl2_bf = (__bf16*)carve(2ull * NCL * H_DIM);
    __bf16* zdr2_bf = (__bf16*)carve(2ull * NDR * H_DIM);

    auto blocks = [](long long n) { return (unsigned)((n + 255) / 256); };

    // ---- pack activations + weights to bf16 (padded / transposed) ----
    pack_a<<<blocks((long long)NCL * FCLP), 256, 0, stream>>>(x_cl, xcl_bf, NCL, FCL, FCLP);
    pack_a<<<blocks((long long)NDR * FDR),  256, 0, stream>>>(x_dr, xdr_bf, NDR, FDR, FDR);

    long long btcl = (long long)H_DIM * FCLP, btdr = (long long)H_DIM * FDR,
              bth  = (long long)H_DIM * H_DIM;
    pack_bt<<<dim3(blocks(btcl), 3), 256, 0, stream>>>(W1s, bt1_cl, FCL, H_DIM, FCLP);
    pack_bt<<<dim3(blocks(btcl), 1), 256, 0, stream>>>(r1r, bt1_cl + 3 * btcl, FCL, H_DIM, FCLP);
    pack_bt<<<dim3(blocks(btdr), 3), 256, 0, stream>>>(W1r, bt1_dr, FDR, H_DIM, FDR);
    pack_bt<<<dim3(blocks(btdr), 1), 256, 0, stream>>>(r1s, bt1_dr + 3 * btdr, FDR, H_DIM, FDR);
    pack_bt<<<dim3(blocks(bth), 3), 256, 0, stream>>>(W2s, bt2_cl, H_DIM, H_DIM, H_DIM);
    pack_bt<<<dim3(blocks(bth), 1), 256, 0, stream>>>(r2r, bt2_cl + 3 * bth, H_DIM, H_DIM, H_DIM);
    pack_bt<<<dim3(blocks(bth), 3), 256, 0, stream>>>(W2r, bt2_dr, H_DIM, H_DIM, H_DIM);
    pack_bt<<<dim3(blocks(bth), 1), 256, 0, stream>>>(r2s, bt2_dr + 3 * bth, H_DIM, H_DIM, H_DIM);
    pack_bt<<<dim3(blocks((long long)H_DIM * 2 * H_DIM), 1), 256, 0, stream>>>(
        l1w, l1t, 2 * H_DIM, H_DIM, 2 * H_DIM);

    // ---- segment counts (identical for both layers) ----
    hipMemsetAsync(ct_cl, 0, sizeof(float) * (size_t)NCL * R_REL, stream);
    hipMemsetAsync(ct_dr, 0, sizeof(float) * (size_t)NDR * R_REL, stream);
    edge_count<<<blocks(NE), 256, 0, stream>>>(dst_sen, et_sen, ct_dr, NE);
    edge_count<<<blocks(NE), 256, 0, stream>>>(dst_rev, et_rev, ct_cl, NE);

    // ---- layer 1: fused (3 relations + root) transforms ----
    wmma_gemm_r4<<<dim3(H_DIM / 64, (NCL + 63) / 64), 128, 0, stream>>>(xcl_bf, bt1_cl, c_cl, NCL, FCLP);
    wmma_gemm_r4<<<dim3(H_DIM / 64, (NDR + 63) / 64), 128, 0, stream>>>(xdr_bf, bt1_dr, c_dr, NDR, FDR);

    hipMemsetAsync(sm_dr, 0, sizeof(float) * (size_t)NDR * R_REL * H_DIM, stream);
    hipMemsetAsync(sm_cl, 0, sizeof(float) * (size_t)NCL * R_REL * H_DIM, stream);
    edge_scatter<<<blocks((long long)NE * 64), 256, 0, stream>>>(c_cl, src_sen, dst_sen, et_sen, sm_dr, NE, NCL);
    edge_scatter<<<blocks((long long)NE * 64), 256, 0, stream>>>(c_dr, src_rev, dst_rev, et_rev, sm_cl, NE, NDR);
    combine<<<blocks((long long)NDR * H_DIM), 256, 0, stream>>>(
        sm_dr, ct_dr, c_dr + 3ull * NDR * H_DIM, b1s, zdr_bf, NDR, 1);
    combine<<<blocks((long long)NCL * H_DIM), 256, 0, stream>>>(
        sm_cl, ct_cl, c_cl + 3ull * NCL * H_DIM, b1r, zcl_bf, NCL, 1);

    // ---- layer 2 (reuse c_cl/c_dr/sm buffers) ----
    wmma_gemm_r4<<<dim3(H_DIM / 64, (NCL + 63) / 64), 128, 0, stream>>>(zcl_bf, bt2_cl, c_cl, NCL, H_DIM);
    wmma_gemm_r4<<<dim3(H_DIM / 64, (NDR + 63) / 64), 128, 0, stream>>>(zdr_bf, bt2_dr, c_dr, NDR, H_DIM);

    hipMemsetAsync(sm_dr, 0, sizeof(float) * (size_t)NDR * R_REL * H_DIM, stream);
    hipMemsetAsync(sm_cl, 0, sizeof(float) * (size_t)NCL * R_REL * H_DIM, stream);
    edge_scatter<<<blocks((long long)NE * 64), 256, 0, stream>>>(c_cl, src_sen, dst_sen, et_sen, sm_dr, NE, NCL);
    edge_scatter<<<blocks((long long)NE * 64), 256, 0, stream>>>(c_dr, src_rev, dst_rev, et_rev, sm_cl, NE, NDR);
    combine<<<blocks((long long)NDR * H_DIM), 256, 0, stream>>>(
        sm_dr, ct_dr, c_dr + 3ull * NDR * H_DIM, b2s, zdr2_bf, NDR, 0);
    combine<<<blocks((long long)NCL * H_DIM), 256, 0, stream>>>(
        sm_cl, ct_cl, c_cl + 3ull * NCL * H_DIM, b2r, zcl2_bf, NCL, 0);

    // ---- fused decoder ----
    init_out<<<blocks(NEL), 256, 0, stream>>>(out, l2b, NEL);
    decoder_wmma<<<dim3(H_DIM / 64, NEL / 16), 128, 0, stream>>>(
        zcl2_bf, zdr2_bf, row, col, l1t, l1b, l2w, out);
}
